// MultiScaleDeformableAttention_70592082477507
// MI455X (gfx1250) — matrix operand order repaired and checked by reading, not compiled, not run on hardware
//
#include <hip/hip_runtime.h>
#include <hip/hip_bf16.h>

typedef __attribute__((ext_vector_type(16))) _Float16 v16h;
typedef __attribute__((ext_vector_type(8)))  _Float16 v8h;
typedef __attribute__((ext_vector_type(8)))  float    v8f;

// ---------------------------------------------------------------------------
// Problem constants (from the reference)
// ---------------------------------------------------------------------------
constexpr int Bc  = 4;
constexpr int Qc  = 5440;
constexpr int Dc  = 256;
constexpr int NHc = 8;
constexpr int NLc = 4;
constexpr int NPc = 4;
constexpr int HDc = 32;   // D / NH
constexpr int Sc  = 5440; // 64*64+32*32+16*16+8*8
constexpr int Mc  = Bc * Qc; // 21760 rows for every GEMM
constexpr int Kc  = 256;     // K for every GEMM

// ---------------------------------------------------------------------------
// GEMM:  C[M,N] = A[M,256] * Bm[256,N] + bias[N]
// f16 WMMA (v_wmma_f32_16x16x32_f16), fp32 accumulate.
// Block = 256 threads (8 waves), tile 64(M) x 64(N), TK = 32.
// Each wave: 16x32 output = two 16x16 WMMA tiles sharing one A fragment.
// Software-pipelined: next K-step's global loads issued before current WMMAs.
// ---------------------------------------------------------------------------
#define TM 64
#define TN 64
#define TK 32
#define LDA 40  // halfs per LDS A row (80B, 16B aligned)
#define LDB 40  // halfs per LDS B^T row

__global__ __launch_bounds__(256)
void msda_gemm_f16wmma(const float* __restrict__ A,
                       const float* __restrict__ Bm,
                       const float* __restrict__ bias,
                       float* __restrict__ C,
                       int N) {
  __shared__ __align__(16) _Float16 As[TM * LDA];      // [row][k]
  __shared__ __align__(16) _Float16 Bt[TN * LDB];      // [n][k] (transposed)

  const int t    = threadIdx.x;
  const int wave = t >> 5;
  const int lane = t & 31;
  const int lh   = lane & 15;        // lane % 16
  const int hi   = lane >> 4;        // 0 or 1
  const int m_tile = wave & 3;       // 0..3   (16 rows each)
  const int n_half = wave >> 2;      // 0..1   (32 cols each)

  const int m0 = blockIdx.y * TM;
  const int n0 = blockIdx.x * TN;

  v8f acc0 = {};
  v8f acc1 = {};

  // staging coordinates: A tile 64x32 (8 f32/thread), B tile 32x64 (8 f32/thread)
  const int a_row = t >> 2;          // 0..63
  const int a_k   = (t & 3) * 8;     // 0,8,16,24
  const int b_k   = t >> 3;          // 0..31
  const int b_n   = (t & 7) * 8;     // 0..56

  const float4* aptr = reinterpret_cast<const float4*>(
      A + (size_t)(m0 + a_row) * Kc + a_k);
  const float4* bptr = reinterpret_cast<const float4*>(
      Bm + (size_t)b_k * N + n0 + b_n);
  const int bstep = (TK * N) >> 2;   // float4 stride for K += 32

  // ---- prologue: load K-step 0 into registers ----
  float4 ra0 = aptr[0], ra1 = aptr[1];
  float4 rb0 = bptr[0], rb1 = bptr[1];

  for (int kb = 0; kb < Kc; kb += TK) {
    // ---- stage current K-step into LDS (f32 -> f16) ----
    {
      _Float16* dst = &As[a_row * LDA + a_k];
      dst[0] = (_Float16)ra0.x; dst[1] = (_Float16)ra0.y;
      dst[2] = (_Float16)ra0.z; dst[3] = (_Float16)ra0.w;
      dst[4] = (_Float16)ra1.x; dst[5] = (_Float16)ra1.y;
      dst[6] = (_Float16)ra1.z; dst[7] = (_Float16)ra1.w;

      Bt[(b_n + 0) * LDB + b_k] = (_Float16)rb0.x;
      Bt[(b_n + 1) * LDB + b_k] = (_Float16)rb0.y;
      Bt[(b_n + 2) * LDB + b_k] = (_Float16)rb0.z;
      Bt[(b_n + 3) * LDB + b_k] = (_Float16)rb0.w;
      Bt[(b_n + 4) * LDB + b_k] = (_Float16)rb1.x;
      Bt[(b_n + 5) * LDB + b_k] = (_Float16)rb1.y;
      Bt[(b_n + 6) * LDB + b_k] = (_Float16)rb1.z;
      Bt[(b_n + 7) * LDB + b_k] = (_Float16)rb1.w;
    }
    __syncthreads();

    // ---- prefetch next K-step from global (overlaps with WMMAs below) ----
    if (kb + TK < Kc) {
      aptr += TK / 4;   // advance 32 floats along K (row stride is Kc)
      bptr += bstep;
      ra0 = aptr[0]; ra1 = aptr[1];
      rb0 = bptr[0]; rb1 = bptr[1];
    }

    // ---- A fragment (16x32): lane<16 row=lh K 0..7 & 16..23 ; lane>=16 +8 ----
    const _Float16* ap = &As[(m_tile * 16 + lh) * LDA + hi * 8];
    v8h alo = *reinterpret_cast<const v8h*>(ap);
    v8h ahi = *reinterpret_cast<const v8h*>(ap + 16);
    v16h af;
#pragma unroll
    for (int i = 0; i < 8; ++i) { af[i] = alo[i]; af[i + 8] = ahi[i]; }

    // ---- two B fragments (32x16 each): col = n_half*32 + s*16 + lh ----
#pragma unroll
    for (int s = 0; s < 2; ++s) {
      const _Float16* bp = &Bt[(n_half * 32 + s * 16 + lh) * LDB + hi * 8];
      v8h blo = *reinterpret_cast<const v8h*>(bp);
      v8h bhi = *reinterpret_cast<const v8h*>(bp + 16);
      v16h bf;
#pragma unroll
      for (int i = 0; i < 8; ++i) { bf[i] = blo[i]; bf[i + 8] = bhi[i]; }

      if (s == 0) {
        acc0 = __builtin_amdgcn_wmma_f32_16x16x32_f16(
            false, af, false, bf, (short)0, acc0, false, false);
      } else {
        acc1 = __builtin_amdgcn_wmma_f32_16x16x32_f16(
            false, af, false, bf, (short)0, acc1, false, false);
      }
    }
    __syncthreads();
  }

  // ---- store: row = m0 + m_tile*16 + 8*hi + i ; col = n0 + n_half*32 + s*16 + lh
  const int rowb = m0 + m_tile * 16 + hi * 8;
  const int col0 = n0 + n_half * 32 + lh;
  const int col1 = col0 + 16;
  const float bv0 = bias[col0];
  const float bv1 = bias[col1];
#pragma unroll
  for (int i = 0; i < 8; ++i) {
    C[(size_t)(rowb + i) * N + col0] = acc0[i] + bv0;
    C[(size_t)(rowb + i) * N + col1] = acc1[i] + bv1;
  }
}

// ---------------------------------------------------------------------------
// Sampling: one wave32 per (b, q, head); 32 lanes = 32 channels (HD).
// Lanes 0..15 own the 16 (level, point) slots: softmax + location math,
// then each slot is broadcast and all lanes gather its 4 bilinear corners
// as coalesced 128B loads (value fits in the 192MB L2).
// ---------------------------------------------------------------------------
__global__ __launch_bounds__(256)
void msda_sample(const float* __restrict__ value,   // [B,S,NH,HD]
                 const float* __restrict__ refp,    // [B,Q,NL,2]
                 const float* __restrict__ offs,    // [B*Q, NH*NL*NP*2]
                 const float* __restrict__ logits,  // [B*Q, NH*NL*NP]
                 const int*   __restrict__ shapes,  // [NL,2]
                 const int*   __restrict__ lstart,  // [NL]
                 float* __restrict__ hout) {        // [B*Q, D]
  const int w    = blockIdx.x * 8 + (threadIdx.x >> 5);
  const int lane = threadIdx.x & 31;
  const int h    = w & (NHc - 1);
  const int bq   = w >> 3;                 // 0 .. B*Q-1
  const int b    = bq / Qc;

  // ---- softmax over the 16 (level,point) slots (lanes 0..15 active) ----
  float logit = -3.0e38f;
  if (lane < 16) logit = logits[(size_t)bq * (NHc * NLc * NPc) + h * 16 + lane];
  float mx = logit;
#pragma unroll
  for (int o = 16; o >= 1; o >>= 1) mx = fmaxf(mx, __shfl_xor(mx, o, 32));
  float e = (lane < 16) ? __expf(logit - mx) : 0.0f;
  float sum = e;
#pragma unroll
  for (int o = 16; o >= 1; o >>= 1) sum += __shfl_xor(sum, o, 32);
  const float wat = e / sum;               // valid on lanes 0..15

  // ---- per-slot sampling location (lanes 0..15) ----
  int   Hl = 1, Wl = 1, base = 0;
  float px = 0.0f, py = 0.0f;
  if (lane < 16) {
    const int lvl = lane >> 2;             // NP = 4
    Hl   = shapes[lvl * 2 + 0];
    Wl   = shapes[lvl * 2 + 1];
    base = lstart[lvl];
    const float rx = refp[((size_t)bq * NLc + lvl) * 2 + 0];
    const float ry = refp[((size_t)bq * NLc + lvl) * 2 + 1];
    const float ox = offs[(size_t)bq * 256 + h * 32 + lane * 2 + 0];
    const float oy = offs[(size_t)bq * 256 + h * 32 + lane * 2 + 1];
    const float lx = rx + ox / (float)Wl;
    const float ly = ry + oy / (float)Hl;
    px = lx * (float)Wl - 0.5f;
    py = ly * (float)Hl - 0.5f;
  }

  float acc = 0.0f;
  const size_t vbase = ((size_t)b * Sc) * Dc + h * HDc + lane;

#pragma unroll
  for (int p = 0; p < 16; ++p) {
    const float x   = __shfl(px,  p, 32);
    const float y   = __shfl(py,  p, 32);
    const float aw  = __shfl(wat, p, 32);
    const int   hl  = __shfl(Hl,  p, 32);
    const int   wl  = __shfl(Wl,  p, 32);
    const int   bs  = __shfl(base, p, 32);

    const float x0f = floorf(x), y0f = floorf(y);
    const float fx  = x - x0f,   fy  = y - y0f;
    const int   x0  = (int)x0f,  y0  = (int)y0f;

#pragma unroll
    for (int dy = 0; dy < 2; ++dy) {
#pragma unroll
      for (int dx = 0; dx < 2; ++dx) {
        const int xi = x0 + dx, yi = y0 + dy;
        const float cw = (dx ? fx : 1.0f - fx) * (dy ? fy : 1.0f - fy);
        const bool valid = (xi >= 0) & (xi < wl) & (yi >= 0) & (yi < hl);
        int xc = xi < 0 ? 0 : (xi > wl - 1 ? wl - 1 : xi);
        int yc = yi < 0 ? 0 : (yi > hl - 1 ? hl - 1 : yi);
        const int idx = bs + yc * wl + xc;
        const float v = value[vbase + (size_t)idx * Dc];
        acc += aw * cw * (valid ? v : 0.0f);
      }
    }
  }

  hout[(size_t)bq * Dc + h * HDc + lane] = acc;
}

// ---------------------------------------------------------------------------
// Launch
// ---------------------------------------------------------------------------
extern "C" void kernel_launch(void* const* d_in, const int* in_sizes, int n_in,
                              void* d_out, int out_size, void* d_ws, size_t ws_size,
                              hipStream_t stream) {
  const float* query  = (const float*)d_in[0];
  const float* refp   = (const float*)d_in[1];
  const float* inpf   = (const float*)d_in[2];
  const int*   shapes = (const int*)  d_in[3];
  const int*   lstart = (const int*)  d_in[4];
  const float* W_off  = (const float*)d_in[5];
  const float* b_off  = (const float*)d_in[6];
  const float* W_attn = (const float*)d_in[7];
  const float* b_attn = (const float*)d_in[8];
  const float* W_val  = (const float*)d_in[9];
  const float* b_val  = (const float*)d_in[10];
  const float* W_out  = (const float*)d_in[11];
  const float* b_out  = (const float*)d_in[12];
  float* out = (float*)d_out;

  float* ws = (float*)d_ws;
  const size_t nBQ = (size_t)Mc;                 // 21760 (== B*S too)
  float* v_val  = ws;                            // [21760, 256]
  float* v_off  = v_val  + nBQ * 256;            // [21760, 256]
  float* v_attn = v_off  + nBQ * 256;            // [21760, 128]
  float* v_hout = v_attn + nBQ * 128;            // [21760, 256]

  const dim3 blk(256);
  const dim3 g256(256 / TN, Mc / TM);            // (4, 340)
  const dim3 g128(128 / TN, Mc / TM);            // (2, 340)

  // 1) value projection
  msda_gemm_f16wmma<<<g256, blk, 0, stream>>>(inpf,  W_val,  b_val,  v_val,  256);
  // 2) sampling offsets
  msda_gemm_f16wmma<<<g256, blk, 0, stream>>>(query, W_off,  b_off,  v_off,  256);
  // 3) attention logits
  msda_gemm_f16wmma<<<g128, blk, 0, stream>>>(query, W_attn, b_attn, v_attn, 128);
  // 4) softmax + bilinear sampling + head accumulation
  msda_sample<<<dim3(Mc * NHc / 8), blk, 0, stream>>>(
      v_val, refp, v_off, v_attn, shapes, lstart, v_hout);
  // 5) output projection
  msda_gemm_f16wmma<<<g256, blk, 0, stream>>>(v_hout, W_out, b_out, out, 256);
}
